// CondAttnBlock_57586921505266
// MI455X (gfx1250) — matrix operand order repaired and checked by reading, not compiled, not run on hardware
//
#include <hip/hip_runtime.h>
#include <hip/hip_bf16.h>

typedef __attribute__((ext_vector_type(16))) _Float16 v16h;
typedef __attribute__((ext_vector_type(8)))  float    v8f;
typedef __attribute__((ext_vector_type(4)))  float    f32x4;

union HF16 { v16h v; _Float16 h[16]; f32x4 q[2]; };

// ---------------------------------------------------------------------------
// WMMA fragment loaders (CDNA5 v_wmma_f32_16x16x32_f16 layouts, ISA 7.12.2)
// A 16x32 (MxK), row-major, ld in elements. lane r=lane&15, hi=lane>>4:
//   elements 0..7  = K: hi*8 + 0..7       (16B contiguous)
//   elements 8..15 = K: 16 + hi*8 + 0..7  (16B contiguous)
__device__ inline v16h ldA(const _Float16* base, int ld, int lane) {
  const int r = lane & 15, hi = lane >> 4;
  const _Float16* p = base + (size_t)r * ld + hi * 8;
  HF16 u;
  u.q[0] = *(const f32x4*)(p);
  u.q[1] = *(const f32x4*)(p + 16);
  return u.v;
}
// B 32x16 (KxN), stored [N][K] row-major (i.e. weight [out][in]).
// lane n=lane&15, hi=lane>>4: elements 0..7 = K: hi*8 + 0..7, 8..15 = K: 16 + hi*8 + 0..7 of column n.
__device__ inline v16h ldB(const _Float16* base, int ld, int lane) {
  const int r = lane & 15, hi = lane >> 4;
  const _Float16* p = base + (size_t)r * ld + hi * 8;
  HF16 u;
  u.q[0] = *(const f32x4*)(p);
  u.q[1] = *(const f32x4*)(p + 16);
  return u.v;
}

__device__ inline v8f wmma16(v16h a, v16h b, v8f c) {
  return __builtin_amdgcn_wmma_f32_16x16x32_f16(false, a, false, b, (short)0, c,
                                                false, false);
}

// Problem constants
#define Bc 32
#define Cc 512
#define Sc 1024   // H*W
#define Mc 256
#define Dc 768

// ---------------------------------------------------------------------------
// fp32 -> fp16 convert
__global__ void cvt_f16_kernel(const float* __restrict__ src,
                               _Float16* __restrict__ dst, int n) {
  int i = blockIdx.x * 256 + threadIdx.x;
  if (i < n) dst[i] = (_Float16)src[i];
}

// ---------------------------------------------------------------------------
// GroupNorm: one block per (b, group). 16 channels x 1024 pixels each.
// Writes h channel-last as f16 [B, S, C].
__global__ void gn_kernel(const float* __restrict__ x,
                          const float* __restrict__ gs,
                          const float* __restrict__ gb,
                          _Float16* __restrict__ h16) {
  const int b = blockIdx.x >> 5, g = blockIdx.x & 31;
  const float* xp = x + ((size_t)(b * Cc + g * 16)) * Sc;  // 16 contiguous chans
  float s = 0.f, s2 = 0.f;
  for (int i = threadIdx.x; i < 16 * Sc; i += 256) {
    float v = xp[i];
    s += v; s2 += v * v;
  }
  __shared__ float rs[256], rs2[256];
  rs[threadIdx.x] = s; rs2[threadIdx.x] = s2;
  __syncthreads();
  for (int o = 128; o > 0; o >>= 1) {
    if (threadIdx.x < o) {
      rs[threadIdx.x]  += rs[threadIdx.x + o];
      rs2[threadIdx.x] += rs2[threadIdx.x + o];
    }
    __syncthreads();
  }
  const float inv_n = 1.f / (16.f * (float)Sc);
  const float mean = rs[0] * inv_n;
  const float var  = rs2[0] * inv_n - mean * mean;
  const float inv  = rsqrtf(var + 1e-5f);
  for (int i = threadIdx.x; i < 16 * Sc; i += 256) {
    const int cl = i >> 10, sp = i & (Sc - 1);
    const int c = g * 16 + cl;
    const float v = (xp[i] - mean) * inv * gs[c] + gb[c];
    h16[((size_t)b * Sc + sp) * Cc + c] = (_Float16)v;
  }
}

// ---------------------------------------------------------------------------
// K/V projection, register-blocked: wave computes 32m x 32c for BOTH k and v
// (shared A fragments). Block = 8 waves (2 m-waves x 4 c-waves) = 64m x 128c.
// Writes k as [B,M,C] and v transposed [B,C,M].
__global__ void kv_kernel(const _Float16* __restrict__ y16,
                          const _Float16* __restrict__ wk16,
                          const _Float16* __restrict__ wv16,
                          const float* __restrict__ bk,
                          const float* __restrict__ bv,
                          _Float16* __restrict__ k16,
                          _Float16* __restrict__ vT) {
  const int lane = threadIdx.x & 31;
  const int wave = threadIdx.x >> 5;
  const int b   = blockIdx.x >> 4;          // 16 blocks per batch
  const int r   = blockIdx.x & 15;
  const int mb  = r >> 2;                   // 4 m-blocks of 64
  const int cb  = r & 3;                    // 4 c-blocks of 128
  const int wr  = wave >> 2, wc = wave & 3;
  const int m0  = mb * 64 + wr * 32;
  const int c0  = cb * 128 + wc * 32;

  const _Float16* aB  = y16  + ((size_t)b * Mc + m0) * Dc;
  const _Float16* kB  = wk16 + (size_t)c0 * Dc;
  const _Float16* vB  = wv16 + (size_t)c0 * Dc;

  v8f aK[2][2] = {}, aV[2][2] = {};
  for (int d0 = 0; d0 < Dc; d0 += 32) {
    v16h a0 = ldA(aB + d0, Dc, lane);
    v16h a1 = ldA(aB + (size_t)16 * Dc + d0, Dc, lane);
    v16h k0 = ldB(kB + d0, Dc, lane);
    v16h k1 = ldB(kB + (size_t)16 * Dc + d0, Dc, lane);
    v16h v0 = ldB(vB + d0, Dc, lane);
    v16h v1 = ldB(vB + (size_t)16 * Dc + d0, Dc, lane);
    aK[0][0] = wmma16(a0, k0, aK[0][0]);
    aK[0][1] = wmma16(a0, k1, aK[0][1]);
    aK[1][0] = wmma16(a1, k0, aK[1][0]);
    aK[1][1] = wmma16(a1, k1, aK[1][1]);
    aV[0][0] = wmma16(a0, v0, aV[0][0]);
    aV[0][1] = wmma16(a0, v1, aV[0][1]);
    aV[1][0] = wmma16(a1, v0, aV[1][0]);
    aV[1][1] = wmma16(a1, v1, aV[1][1]);
  }
  const int col = lane & 15, hi = lane >> 4;
#pragma unroll
  for (int i = 0; i < 2; ++i) {
#pragma unroll
    for (int t = 0; t < 2; ++t) {
      const int c = c0 + t * 16 + col;
      const float kbias = bk[c], vbias = bv[c];
      _Float16 tmp[8];
#pragma unroll
      for (int j = 0; j < 8; ++j) {
        const int m = m0 + i * 16 + hi * 8 + j;
        k16[((size_t)b * Mc + m) * Cc + c] = (_Float16)(aK[i][t][j] + kbias);
        tmp[j] = (_Float16)(aV[i][t][j] + vbias);
      }
      _Float16* dst = vT + ((size_t)b * Cc + c) * Mc + m0 + i * 16 + hi * 8;
      *(f32x4*)dst = *(const f32x4*)tmp;                 // contiguous 16B store
    }
  }
}

// ---------------------------------------------------------------------------
// Q projection, register-blocked: wave tile 32s x 64c (8 accumulators,
// 2 A-frags + 4 B-frags per K-step). Block = 8 waves (4x2) = 128s x 128c.
// q = (h.wq^T + bq) * C^-0.5, stored f16 [B,S,C].
__global__ void qproj_kernel(const _Float16* __restrict__ h16,
                             const _Float16* __restrict__ wq16,
                             const float* __restrict__ bq,
                             _Float16* __restrict__ q16) {
  const int lane = threadIdx.x & 31;
  const int wave = threadIdx.x >> 5;
  const int b  = blockIdx.x >> 5;           // 32 blocks per batch
  const int r  = blockIdx.x & 31;
  const int rb = r >> 2;                    // 8 row-blocks of 128
  const int cb = r & 3;                     // 4 col-blocks of 128
  const int wr = wave >> 1, wc = wave & 1;
  const int s0 = rb * 128 + wr * 32;
  const int c0 = cb * 128 + wc * 64;

  const _Float16* aB = h16  + ((size_t)b * Sc + s0) * Cc;
  const _Float16* bB = wq16 + (size_t)c0 * Cc;

  v8f acc[2][4] = {};
  for (int k0 = 0; k0 < Cc; k0 += 32) {
    v16h a0 = ldA(aB + k0, Cc, lane);
    v16h a1 = ldA(aB + (size_t)16 * Cc + k0, Cc, lane);
    v16h bf[4];
#pragma unroll
    for (int t = 0; t < 4; ++t)
      bf[t] = ldB(bB + (size_t)(t * 16) * Cc + k0, Cc, lane);
#pragma unroll
    for (int t = 0; t < 4; ++t) {
      acc[0][t] = wmma16(a0, bf[t], acc[0][t]);
      acc[1][t] = wmma16(a1, bf[t], acc[1][t]);
    }
  }
  const int col = lane & 15, hi = lane >> 4;
  const float scale = 0.044194173824159216f;  // 512^-0.5 folded into q
#pragma unroll
  for (int i = 0; i < 2; ++i) {
#pragma unroll
    for (int t = 0; t < 4; ++t) {
      const int c = c0 + t * 16 + col;
      const float bias = bq[c];
#pragma unroll
      for (int j = 0; j < 8; ++j) {
        const int s = s0 + i * 16 + hi * 8 + j;
        q16[((size_t)b * Sc + s) * Cc + c] = (_Float16)((acc[i][t][j] + bias) * scale);
      }
    }
  }
}

// ---------------------------------------------------------------------------
// Fused attention + output projection + residual. One block per (b, 32-row
// s-tile); 8 waves. scores -> softmax -> w.v -> proj all via WMMA, with the
// softmax probabilities and attention output kept in LDS as A-operands.
__global__ void attn_kernel(const _Float16* __restrict__ q16,
                            const _Float16* __restrict__ k16,
                            const _Float16* __restrict__ vT,
                            const _Float16* __restrict__ wp16,
                            const float* __restrict__ bp,
                            const float* __restrict__ x,
                            float* __restrict__ out) {
  const int b  = blockIdx.x >> 5;   // 32 s-tiles (of 32 rows) per batch
  const int sb = blockIdx.x & 31;
  const int tid = threadIdx.x;
  const int wave = tid >> 5, lane = tid & 31;
  const int col = lane & 15, hi = lane >> 4;

  __shared__ float     sc[32 * Mc];       // f32 scores 32x256 (32 KB)
  __shared__ float     red[32 * 9];       // row reductions
  __shared__ _Float16  pr[32 * Mc];       // f16 probs  32x256 (16 KB, A operand)
  __shared__ _Float16  hs[32 * Cc];       // f16 attn-h 32x512 (32 KB, A operand)

  const _Float16* qb = q16 + ((size_t)b * Sc + sb * 32) * Cc;
  const _Float16* kb = k16 + (size_t)b * Mc * Cc;

  // warm L1 with this wave's projection-weight columns (global_prefetch_b8)
#pragma unroll
  for (int t = 0; t < 4; ++t)
    __builtin_prefetch(wp16 + (size_t)(wave * 64 + t * 16 + col) * Cc, 0, 1);

  // ---- scores: wave w owns m-columns [w*32, w*32+32), both 16-row halves
  {
    const int n0 = wave * 32;
    v8f a00 = {}, a01 = {}, a10 = {}, a11 = {};
    for (int kk = 0; kk < Cc; kk += 32) {
      v16h a0 = ldA(qb + kk, Cc, lane);
      v16h a1 = ldA(qb + (size_t)16 * Cc + kk, Cc, lane);
      v16h b0 = ldB(kb + (size_t)n0 * Cc + kk, Cc, lane);
      v16h b1 = ldB(kb + (size_t)(n0 + 16) * Cc + kk, Cc, lane);
      a00 = wmma16(a0, b0, a00);
      a01 = wmma16(a0, b1, a01);
      a10 = wmma16(a1, b0, a10);
      a11 = wmma16(a1, b1, a11);
    }
#pragma unroll
    for (int j = 0; j < 8; ++j) {
      sc[(hi * 8 + j) * Mc + n0 + col]           = a00[j];
      sc[(hi * 8 + j) * Mc + n0 + 16 + col]      = a01[j];
      sc[(16 + hi * 8 + j) * Mc + n0 + col]      = a10[j];
      sc[(16 + hi * 8 + j) * Mc + n0 + 16 + col] = a11[j];
    }
  }
  __syncthreads();

  // ---- softmax over m (256) per row; 8 threads per row x 32 cols each
  {
    const int r = tid & 31, ch = tid >> 5;
    float m = -3.0e38f;
    for (int j = 0; j < 32; ++j) m = fmaxf(m, sc[r * Mc + ch * 32 + j]);
    red[r * 9 + ch] = m;
    __syncthreads();
    if (ch == 0) {
      float mm = red[r * 9];
      for (int j = 1; j < 8; ++j) mm = fmaxf(mm, red[r * 9 + j]);
      red[r * 9 + 8] = mm;
    }
    __syncthreads();
    const float rm = red[r * 9 + 8];
    float s = 0.f;
    for (int j = 0; j < 32; ++j) {
      const float e = __expf(sc[r * Mc + ch * 32 + j] - rm);
      sc[r * Mc + ch * 32 + j] = e;
      s += e;
    }
    red[r * 9 + ch] = s;
    __syncthreads();
    if (ch == 0) {
      float ss = 0.f;
      for (int j = 0; j < 8; ++j) ss += red[r * 9 + j];
      red[r * 9 + 8] = ss;
    }
    __syncthreads();
    const float inv = 1.0f / red[r * 9 + 8];
    for (int j = 0; j < 32; ++j)
      pr[r * Mc + ch * 32 + j] = (_Float16)(sc[r * Mc + ch * 32 + j] * inv);
  }
  __syncthreads();

  // ---- h = probs @ v : wave w owns c-cols [w*64, +64), 2x4 register block
  {
    const _Float16* vb = vT + (size_t)b * Cc * Mc;
    v8f acc[2][4] = {};
    for (int mm = 0; mm < Mc; mm += 32) {
      v16h a0 = ldA(pr + mm, Mc, lane);                        // LDS A
      v16h a1 = ldA(pr + 16 * Mc + mm, Mc, lane);
      v16h bf[4];
#pragma unroll
      for (int t = 0; t < 4; ++t)
        bf[t] = ldB(vb + (size_t)(wave * 64 + t * 16) * Mc + mm, Mc, lane);
#pragma unroll
      for (int t = 0; t < 4; ++t) {
        acc[0][t] = wmma16(a0, bf[t], acc[0][t]);
        acc[1][t] = wmma16(a1, bf[t], acc[1][t]);
      }
    }
#pragma unroll
    for (int i = 0; i < 2; ++i)
#pragma unroll
      for (int t = 0; t < 4; ++t)
#pragma unroll
        for (int j = 0; j < 8; ++j)
          hs[(i * 16 + hi * 8 + j) * Cc + wave * 64 + t * 16 + col] =
              (_Float16)acc[i][t][j];
  }
  __syncthreads();

  // ---- out = x + h @ wp^T + bp : wave w owns o-cols [w*64, +64), 2x4 block
  {
    v8f acc[2][4] = {};
    for (int cc = 0; cc < Cc; cc += 32) {
      v16h a0 = ldA(hs + cc, Cc, lane);                        // LDS A
      v16h a1 = ldA(hs + 16 * Cc + cc, Cc, lane);
      v16h bf[4];
#pragma unroll
      for (int t = 0; t < 4; ++t)
        bf[t] = ldB(wp16 + (size_t)(wave * 64 + t * 16) * Cc + cc, Cc, lane);
#pragma unroll
      for (int t = 0; t < 4; ++t) {
        acc[0][t] = wmma16(a0, bf[t], acc[0][t]);
        acc[1][t] = wmma16(a1, bf[t], acc[1][t]);
      }
    }
#pragma unroll
    for (int i = 0; i < 2; ++i) {
#pragma unroll
      for (int t = 0; t < 4; ++t) {
        const int o = wave * 64 + t * 16 + col;
        const float bias = bp[o];
#pragma unroll
        for (int j = 0; j < 8; ++j) {
          const int s = sb * 32 + i * 16 + hi * 8 + j;
          const size_t idx = ((size_t)(b * Cc + o)) * Sc + s;
          out[idx] = x[idx] + acc[i][t][j] + bias;
        }
      }
    }
  }
}

// ---------------------------------------------------------------------------
extern "C" void kernel_launch(void* const* d_in, const int* in_sizes, int n_in,
                              void* d_out, int out_size, void* d_ws, size_t ws_size,
                              hipStream_t stream) {
  (void)in_sizes; (void)n_in; (void)out_size; (void)ws_size;

  const float* x  = (const float*)d_in[0];   // [32,512,32,32]
  const float* y  = (const float*)d_in[1];   // [32,256,768]
  const float* gs = (const float*)d_in[2];
  const float* gb = (const float*)d_in[3];
  const float* wq = (const float*)d_in[4];   // [512,512]
  const float* bq = (const float*)d_in[5];
  const float* wk = (const float*)d_in[6];   // [512,768]
  const float* bk = (const float*)d_in[7];
  const float* wv = (const float*)d_in[8];   // [512,768]
  const float* bv = (const float*)d_in[9];
  const float* wp = (const float*)d_in[10];  // [512,512]
  const float* bp = (const float*)d_in[11];
  float* out = (float*)d_out;

  // Workspace carve-up (bytes)
  char* ws = (char*)d_ws;
  size_t off = 0;
  _Float16* y16  = (_Float16*)(ws + off); off += (size_t)Bc * Mc * Dc * 2;  // 12 MB
  _Float16* wq16 = (_Float16*)(ws + off); off += (size_t)Cc * Cc * 2;
  _Float16* wk16 = (_Float16*)(ws + off); off += (size_t)Cc * Dc * 2;
  _Float16* wv16 = (_Float16*)(ws + off); off += (size_t)Cc * Dc * 2;
  _Float16* wp16 = (_Float16*)(ws + off); off += (size_t)Cc * Cc * 2;
  _Float16* h16  = (_Float16*)(ws + off); off += (size_t)Bc * Sc * Cc * 2;  // 32 MB
  _Float16* q16  = (_Float16*)(ws + off); off += (size_t)Bc * Sc * Cc * 2;  // 32 MB
  _Float16* k16  = (_Float16*)(ws + off); off += (size_t)Bc * Mc * Cc * 2;  // 8 MB
  _Float16* vT   = (_Float16*)(ws + off); off += (size_t)Bc * Cc * Mc * 2;  // 8 MB

  // 1) fp32 -> fp16 conversions
  {
    const int ny = Bc * Mc * Dc;
    cvt_f16_kernel<<<(ny + 255) / 256, 256, 0, stream>>>(y, y16, ny);
    cvt_f16_kernel<<<(Cc * Cc + 255) / 256, 256, 0, stream>>>(wq, wq16, Cc * Cc);
    cvt_f16_kernel<<<(Cc * Dc + 255) / 256, 256, 0, stream>>>(wk, wk16, Cc * Dc);
    cvt_f16_kernel<<<(Cc * Dc + 255) / 256, 256, 0, stream>>>(wv, wv16, Cc * Dc);
    cvt_f16_kernel<<<(Cc * Cc + 255) / 256, 256, 0, stream>>>(wp, wp16, Cc * Cc);
  }
  // 2) GroupNorm -> h16 [B,S,C]
  gn_kernel<<<Bc * 32, 256, 0, stream>>>(x, gs, gb, h16);
  // 3) K/V projection: 512 blocks x 8 waves, 64m x 128c block tile
  kv_kernel<<<512, 256, 0, stream>>>(y16, wk16, wv16, bk, bv, k16, vT);
  // 4) Q projection: 1024 blocks x 8 waves, 128s x 128c block tile
  qproj_kernel<<<1024, 256, 0, stream>>>(h16, wq16, bq, q16);
  // 5) fused attention + projection + residual: (b, 32-row tile) per block
  attn_kernel<<<Bc * 32, 256, 0, stream>>>(q16, k16, vT, wp16, bp, x, out);
}
